// EdgewiseMSA_36034775613441
// MI455X (gfx1250) — matrix operand order repaired and checked by reading, not compiled, not run on hardware
//
#include <hip/hip_runtime.h>
#include <math.h>

#define DEV __device__ __forceinline__

typedef __attribute__((ext_vector_type(16))) __bf16 v16bf;
typedef __attribute__((ext_vector_type(8)))  float v8f;

namespace {
constexpr int BB  = 2;
constexpr int NN1 = 1024;          // N
constexpr int DIM = 512;
constexpr int HH  = 8;
constexpr int DK  = 64;
constexpr int HID = 16;
constexpr int BH  = BB * HH;        // 16
constexpr int BN  = BB * NN1;       // 2048
constexpr size_t NSQ = (size_t)NN1 * NN1;   // 1M
constexpr float EPSF = 1e-6f;
constexpr float BETA_NOT = 0.5f;
}

DEV unsigned short f2bf(float f) {
    unsigned int u = __float_as_uint(f);
    unsigned int r = (u + 0x7FFFu + ((u >> 16) & 1u)) >> 16;
    return (unsigned short)r;
}

DEV v16bf load_frag16(const unsigned short* p0, const unsigned short* p1) {
    struct P { uint4 a, b; } p;
    p.a = *reinterpret_cast<const uint4*>(p0);
    p.b = *reinterpret_cast<const uint4*>(p1);
    return __builtin_bit_cast(v16bf, p);
}

DEV v8f wmma_bf16(v16bf a, v16bf b, v8f c) {
    return __builtin_amdgcn_wmma_f32_16x16x32_bf16(false, a, false, b, (short)0, c, false, false);
}

DEV float gelu_tanh(float x) {
    float x3 = x * x * x;
    return 0.5f * x * (1.0f + tanhf(0.7978845608028654f * (x + 0.044715f * x3)));
}
DEV float sigm(float x) { return 1.0f / (1.0f + expf(-x)); }

// ---------------------------------------------------------------------------
// Elementwise casts / transposes (prep)
// ---------------------------------------------------------------------------
__global__ void k_cast_bf16(const float* __restrict__ src, unsigned short* __restrict__ dst, int n) {
    int i = blockIdx.x * blockDim.x + threadIdx.x;
    if (i < n) dst[i] = f2bf(src[i]);
}

__global__ void k_transpose_bf16(const float* __restrict__ src, unsigned short* __restrict__ dst,
                                 int rows, int cols) {
    // src: rows x cols ; dst: cols x rows
    int i = blockIdx.x * blockDim.x + threadIdx.x;
    if (i < rows * cols) {
        int r = i / cols, c = i % cols;
        dst[(size_t)c * rows + r] = f2bf(src[i]);
    }
}

// ---------------------------------------------------------------------------
// QKV projection GEMM: [BN x 1536] = xb[BN x 512] * WT[1536 x 512]^T
// Scatter into q/k (row-major per head) and v^T (DK x N per head).
// ---------------------------------------------------------------------------
__global__ void k_qkv(const unsigned short* __restrict__ xb,
                      const unsigned short* __restrict__ WT0,
                      const unsigned short* __restrict__ WT1,
                      unsigned short* __restrict__ q0, unsigned short* __restrict__ k0,
                      unsigned short* __restrict__ v0T,
                      unsigned short* __restrict__ q1, unsigned short* __restrict__ k1,
                      unsigned short* __restrict__ v1T) {
    const int lane = threadIdx.x & 31;
    const int half = lane >> 4;
    const int l15  = lane & 15;
    const int tilesM = BN / 16, tilesO = (3 * DIM) / 16;
    int wave = blockIdx.x * (blockDim.x >> 5) + (threadIdx.x >> 5);
    int total = 2 * tilesM * tilesO;
    if (wave >= total) return;
    int set = wave / (tilesM * tilesO);
    int rem = wave % (tilesM * tilesO);
    int tm = rem / tilesO, to = rem % tilesO;
    const unsigned short* WT = set ? WT1 : WT0;
    const unsigned short* arow = xb + (size_t)(tm * 16 + l15) * DIM;
    const unsigned short* brow = WT + (size_t)(to * 16 + l15) * DIM;
    const int ka0 = 8 * half, ka1 = 16 + 8 * half;
    const int kb0 = 8 * half, kb1 = 16 + 8 * half;
    v8f acc = {};
    for (int kb = 0; kb < DIM; kb += 32) {
        v16bf av = load_frag16(arow + kb + ka0, arow + kb + ka1);
        v16bf bv = load_frag16(brow + kb + kb0, brow + kb + kb1);
        acc = wmma_bf16(av, bv, acc);
    }
    int o = to * 16 + l15;                // output column in [0, 1536)
    int which = o / DIM;                  // 0=q, 1=k, 2=v
    int hh = (o % DIM) / DK;
    int d  = o % DK;
    unsigned short* qd = set ? q1 : q0;
    unsigned short* kd = set ? k1 : k0;
    unsigned short* vd = set ? v1T : v0T;
#pragma unroll
    for (int r = 0; r < 8; r++) {
        int n  = tm * 16 + r + 8 * half;  // global row = b*N + token
        int b  = n >> 10;
        int nn = n & (NN1 - 1);
        int bh = b * HH + hh;
        float val = acc[r];
        if (which == 0)      qd[((size_t)bh * NN1 + nn) * DK + d] = f2bf(val * 0.125f); // fold 1/sqrt(DK)
        else if (which == 1) kd[((size_t)bh * NN1 + nn) * DK + d] = f2bf(val);
        else                 vd[((size_t)bh * DK + d) * NN1 + nn] = f2bf(val);
    }
}

// ---------------------------------------------------------------------------
// Generic batched WMMA GEMM, f32 output, 2x4 register-blocked (32x64 per wave):
//   C[b] = A[b] (M x K, row-major bf16) * BT[b]^T  (BT is Ncol x K, row-major)
// Requires M % 32 == 0, Ncol % 64 == 0, K % 32 == 0.
// ---------------------------------------------------------------------------
__global__ void k_gemm_bf16_f32(const unsigned short* __restrict__ A,
                                const unsigned short* __restrict__ BT,
                                float* __restrict__ C,
                                int M, int Ncol, int K, int batch,
                                size_t sA, size_t sB, size_t sC) {
    const int lane = threadIdx.x & 31;
    const int half = lane >> 4;
    const int l15  = lane & 15;
    int wave = blockIdx.x * (blockDim.x >> 5) + (threadIdx.x >> 5);
    int tM = M >> 5, tN = Ncol >> 6;
    int total = batch * tM * tN;
    if (wave >= total) return;
    int b = wave / (tM * tN);
    int r = wave % (tM * tN);
    int tm = r / tN, tn = r % tN;
    const unsigned short* arow0 = A + (size_t)b * sA + (size_t)(tm * 32 + l15) * K;
    const unsigned short* arow1 = arow0 + (size_t)16 * K;
    const unsigned short* brow0 = BT + (size_t)b * sB + (size_t)(tn * 64 + l15) * K;
    const unsigned short* brow1 = brow0 + (size_t)16 * K;
    const unsigned short* brow2 = brow0 + (size_t)32 * K;
    const unsigned short* brow3 = brow0 + (size_t)48 * K;
    const int ka0 = 8 * half, ka1 = 16 + 8 * half;
    const int kb0 = 8 * half, kb1 = 16 + 8 * half;
    v8f a00 = {}, a01 = {}, a02 = {}, a03 = {};
    v8f a10 = {}, a11 = {}, a12 = {}, a13 = {};
    for (int kk = 0; kk < K; kk += 32) {
        v16bf fa0 = load_frag16(arow0 + kk + ka0, arow0 + kk + ka1);
        v16bf fa1 = load_frag16(arow1 + kk + ka0, arow1 + kk + ka1);
        v16bf fb0 = load_frag16(brow0 + kk + kb0, brow0 + kk + kb1);
        v16bf fb1 = load_frag16(brow1 + kk + kb0, brow1 + kk + kb1);
        v16bf fb2 = load_frag16(brow2 + kk + kb0, brow2 + kk + kb1);
        v16bf fb3 = load_frag16(brow3 + kk + kb0, brow3 + kk + kb1);
        a00 = wmma_bf16(fa0, fb0, a00);
        a01 = wmma_bf16(fa0, fb1, a01);
        a02 = wmma_bf16(fa0, fb2, a02);
        a03 = wmma_bf16(fa0, fb3, a03);
        a10 = wmma_bf16(fa1, fb0, a10);
        a11 = wmma_bf16(fa1, fb1, a11);
        a12 = wmma_bf16(fa1, fb2, a12);
        a13 = wmma_bf16(fa1, fb3, a13);
    }
    float* Cb = C + (size_t)b * sC;
    auto store_tile = [&](const v8f& acc, int mBase, int nBase) {
        float* crow = Cb + (size_t)mBase * Ncol + nBase + l15;
#pragma unroll
        for (int rr = 0; rr < 8; rr++) crow[(size_t)(rr + 8 * half) * Ncol] = acc[rr];
    };
    store_tile(a00, tm * 32,      tn * 64);
    store_tile(a01, tm * 32,      tn * 64 + 16);
    store_tile(a02, tm * 32,      tn * 64 + 32);
    store_tile(a03, tm * 32,      tn * 64 + 48);
    store_tile(a10, tm * 32 + 16, tn * 64);
    store_tile(a11, tm * 32 + 16, tn * 64 + 16);
    store_tile(a12, tm * 32 + 16, tn * 64 + 32);
    store_tile(a13, tm * 32 + 16, tn * 64 + 48);
}

// ---------------------------------------------------------------------------
// Row softmax over 1024 columns; writes bf16 probs and (optionally) transpose.
// One 256-thread block per row.
// ---------------------------------------------------------------------------
__global__ void k_softmax(const float* __restrict__ S, unsigned short* __restrict__ A,
                          unsigned short* __restrict__ AT) {
    int row = blockIdx.x;                 // [0, BH*N)
    int bh = row >> 10;
    int n  = row & (NN1 - 1);
    const float* srow = S + (size_t)bh * NSQ + (size_t)n * NN1;
    int tid = threadIdx.x;
    __shared__ float red[256];
    float v[4];
    float mx = -3.0e38f;
#pragma unroll
    for (int p = 0; p < 4; p++) { v[p] = srow[tid + p * 256]; mx = fmaxf(mx, v[p]); }
    red[tid] = mx; __syncthreads();
    for (int s = 128; s > 0; s >>= 1) { if (tid < s) red[tid] = fmaxf(red[tid], red[tid + s]); __syncthreads(); }
    mx = red[0]; __syncthreads();
    float sum = 0.f;
#pragma unroll
    for (int p = 0; p < 4; p++) { v[p] = expf(v[p] - mx); sum += v[p]; }
    red[tid] = sum; __syncthreads();
    for (int s = 128; s > 0; s >>= 1) { if (tid < s) red[tid] += red[tid + s]; __syncthreads(); }
    float inv = 1.0f / red[0];
#pragma unroll
    for (int p = 0; p < 4; p++) {
        int m = tid + p * 256;
        unsigned short bv = f2bf(v[p] * inv);
        A[(size_t)bh * NSQ + (size_t)n * NN1 + m] = bv;
        if (AT) AT[(size_t)bh * NSQ + (size_t)m * NN1 + n] = bv;
    }
}

// ---------------------------------------------------------------------------
// Fused gate MLP + Smix. One block handles a 32x32 tile of one (bh) slice.
// Transposed S reads staged through LDS mirror tiles. Smix overwrites Cb.
// ---------------------------------------------------------------------------
__global__ void k_gates(const float* __restrict__ S0, const float* __restrict__ S1,
                        const float* __restrict__ Cf, float* __restrict__ Cb,
                        const float* __restrict__ w1, const float* __restrict__ b1,
                        const float* __restrict__ w2, const float* __restrict__ b2) {
    int blk = blockIdx.x;
    int bh = blk / (32 * 32);
    int rr = blk % (32 * 32);
    int tn = rr / 32, tm = rr % 32;
    int n0 = tn * 32, m0 = tm * 32;
    size_t base = (size_t)bh * NSQ;
    int tid = threadIdx.x;

    __shared__ float L0[32][33];
    __shared__ float L1[32][33];
    __shared__ float sw1[HID * 6], sb1[HID], sw2[4 * HID], sb2[4];
    if (tid < HID * 6) sw1[tid] = w1[tid];
    if (tid < HID)     sb1[tid] = b1[tid];
    if (tid < 4 * HID) sw2[tid] = w2[tid];
    if (tid < 4)       sb2[tid] = b2[tid];
#pragma unroll
    for (int p = 0; p < 4; p++) {
        int e = p * 256 + tid;
        int dn = e & 31, dm = e >> 5;     // mirror tile: rows over m, cols over n (coalesced in dn)
        L0[dm][dn] = S0[base + (size_t)(m0 + dm) * NN1 + n0 + dn];
        L1[dm][dn] = S1[base + (size_t)(m0 + dm) * NN1 + n0 + dn];
    }
    __syncthreads();

#pragma unroll
    for (int p = 0; p < 4; p++) {
        int e = p * 256 + tid;
        int dm = e & 31, dn = e >> 5;     // element (n0+dn, m0+dm); m fastest -> coalesced
        size_t idx = base + (size_t)(n0 + dn) * NN1 + (m0 + dm);
        float s0  = S0[idx];
        float s1  = S1[idx];
        float s0t = L0[dm][dn];
        float s1t = L1[dm][dn];
        float cr  = logf(Cf[idx] + EPSF);
        float cl  = logf(Cb[idx] + EPSF);
        float feat[6] = { s0, s1, s0t, s1t, cr, cl };
        float h1[HID];
#pragma unroll
        for (int o = 0; o < HID; o++) {
            float t = sb1[o];
#pragma unroll
            for (int c = 0; c < 6; c++) t += feat[c] * sw1[o * 6 + c];
            h1[o] = gelu_tanh(t);
        }
        float g[4];
#pragma unroll
        for (int o = 0; o < 4; o++) {
            float t = sb2[o];
#pragma unroll
            for (int c = 0; c < HID; c++) t += h1[c] * sw2[o * HID + c];
            g[o] = sigm(t);
        }
        float mxv = fmaxf(s0, s1);
        float lse = mxv + log1pf(expf(fminf(s0, s1) - mxv));
        float smix = s0 + g[0] * s1 + g[1] * (lse - s0) - g[2] * (BETA_NOT * s1) + g[3] * cr;
        Cb[idx] = smix;                   // in-place, same element
    }
}

// ---------------------------------------------------------------------------
// t = A1 @ v1 per head, stored transposed: tT[bh][d][n], bf16. 1x4 blocked
// (one wave covers the full DK=64 column extent).
// ---------------------------------------------------------------------------
__global__ void k_tchain(const unsigned short* __restrict__ A1,
                         const unsigned short* __restrict__ v1T,
                         unsigned short* __restrict__ tT) {
    const int lane = threadIdx.x & 31;
    const int half = lane >> 4;
    const int l15  = lane & 15;
    int wave = blockIdx.x * (blockDim.x >> 5) + (threadIdx.x >> 5);
    int tM = NN1 / 16;
    int total = BH * tM;
    if (wave >= total) return;
    int bh = wave / tM;
    int tm = wave % tM;
    const unsigned short* arow = A1 + (size_t)bh * NSQ + (size_t)(tm * 16 + l15) * NN1;
    const unsigned short* brow0 = v1T + (size_t)bh * DK * NN1 + (size_t)l15 * NN1;
    const unsigned short* brow1 = brow0 + (size_t)16 * NN1;
    const unsigned short* brow2 = brow0 + (size_t)32 * NN1;
    const unsigned short* brow3 = brow0 + (size_t)48 * NN1;
    const int ka0 = 8 * half, ka1 = 16 + 8 * half;
    const int kb0 = 8 * half, kb1 = 16 + 8 * half;
    v8f c0 = {}, c1 = {}, c2 = {}, c3 = {};
    for (int kk = 0; kk < NN1; kk += 32) {
        v16bf av = load_frag16(arow + kk + ka0, arow + kk + ka1);
        c0 = wmma_bf16(av, load_frag16(brow0 + kk + kb0, brow0 + kk + kb1), c0);
        c1 = wmma_bf16(av, load_frag16(brow1 + kk + kb0, brow1 + kk + kb1), c1);
        c2 = wmma_bf16(av, load_frag16(brow2 + kk + kb0, brow2 + kk + kb1), c2);
        c3 = wmma_bf16(av, load_frag16(brow3 + kk + kb0, brow3 + kk + kb1), c3);
    }
    unsigned short* tbase = tT + (size_t)bh * DK * NN1;
    const v8f* accs[4] = { &c0, &c1, &c2, &c3 };
#pragma unroll
    for (int s = 0; s < 4; s++) {
        int dcol = s * 16 + l15;
#pragma unroll
        for (int rr = 0; rr < 8; rr++) {
            int n = tm * 16 + rr + 8 * half;
            tbase[(size_t)dcol * NN1 + n] = f2bf((*accs[s])[rr]);
        }
    }
}

// ---------------------------------------------------------------------------
// y = Amix @ v0 + sigmoid(chain) * (A0 @ t); bf16 output (B, N, DIM).
// 1x4 blocked, two WMMA chains (8 accumulators).
// ---------------------------------------------------------------------------
__global__ void k_y(const unsigned short* __restrict__ Am, const unsigned short* __restrict__ v0T,
                    const unsigned short* __restrict__ A0, const unsigned short* __restrict__ tT,
                    const float* __restrict__ chain_logit, unsigned short* __restrict__ y) {
    const int lane = threadIdx.x & 31;
    const int half = lane >> 4;
    const int l15  = lane & 15;
    int wave = blockIdx.x * (blockDim.x >> 5) + (threadIdx.x >> 5);
    int tM = NN1 / 16;
    int total = BH * tM;
    if (wave >= total) return;
    int bh = wave / tM;
    int tm = wave % tM;
    const unsigned short* a1row = Am + (size_t)bh * NSQ + (size_t)(tm * 16 + l15) * NN1;
    const unsigned short* a2row = A0 + (size_t)bh * NSQ + (size_t)(tm * 16 + l15) * NN1;
    const unsigned short* p0 = v0T + (size_t)bh * DK * NN1 + (size_t)l15 * NN1;
    const unsigned short* q0 = tT + (size_t)bh * DK * NN1 + (size_t)l15 * NN1;
    const int ka0 = 8 * half, ka1 = 16 + 8 * half;
    const int kb0 = 8 * half, kb1 = 16 + 8 * half;
    v8f u0 = {}, u1 = {}, u2 = {}, u3 = {};
    v8f w0 = {}, w1 = {}, w2 = {}, w3 = {};
    for (int kk = 0; kk < NN1; kk += 32) {
        v16bf av1 = load_frag16(a1row + kk + ka0, a1row + kk + ka1);
        u0 = wmma_bf16(av1, load_frag16(p0 + kk + kb0,                      p0 + kk + kb1), u0);
        u1 = wmma_bf16(av1, load_frag16(p0 + (size_t)16 * NN1 + kk + kb0,   p0 + (size_t)16 * NN1 + kk + kb1), u1);
        u2 = wmma_bf16(av1, load_frag16(p0 + (size_t)32 * NN1 + kk + kb0,   p0 + (size_t)32 * NN1 + kk + kb1), u2);
        u3 = wmma_bf16(av1, load_frag16(p0 + (size_t)48 * NN1 + kk + kb0,   p0 + (size_t)48 * NN1 + kk + kb1), u3);
        v16bf av2 = load_frag16(a2row + kk + ka0, a2row + kk + ka1);
        w0 = wmma_bf16(av2, load_frag16(q0 + kk + kb0,                      q0 + kk + kb1), w0);
        w1 = wmma_bf16(av2, load_frag16(q0 + (size_t)16 * NN1 + kk + kb0,   q0 + (size_t)16 * NN1 + kk + kb1), w1);
        w2 = wmma_bf16(av2, load_frag16(q0 + (size_t)32 * NN1 + kk + kb0,   q0 + (size_t)32 * NN1 + kk + kb1), w2);
        w3 = wmma_bf16(av2, load_frag16(q0 + (size_t)48 * NN1 + kk + kb0,   q0 + (size_t)48 * NN1 + kk + kb1), w3);
    }
    float sig = sigm(chain_logit[0]);
    int b  = bh / HH;
    int hh = bh % HH;
    const v8f* ua[4] = { &u0, &u1, &u2, &u3 };
    const v8f* wa[4] = { &w0, &w1, &w2, &w3 };
#pragma unroll
    for (int s = 0; s < 4; s++) {
        int dc = s * 16 + l15;
#pragma unroll
        for (int rr = 0; rr < 8; rr++) {
            int n = tm * 16 + rr + 8 * half;
            float val = (*ua[s])[rr] + sig * (*wa[s])[rr];
            y[((size_t)b * NN1 + n) * DIM + hh * DK + dc] = f2bf(val);
        }
    }
}

// ---------------------------------------------------------------------------
extern "C" void kernel_launch(void* const* d_in, const int* in_sizes, int n_in,
                              void* d_out, int out_size, void* d_ws, size_t ws_size,
                              hipStream_t stream) {
    const float* x     = (const float*)d_in[0];
    const float* Wq0   = (const float*)d_in[1];
    const float* Wq1   = (const float*)d_in[2];
    const float* Wp    = (const float*)d_in[3];
    const float* c1w   = (const float*)d_in[4];
    const float* c1b   = (const float*)d_in[5];
    const float* c2w   = (const float*)d_in[6];
    const float* c2b   = (const float*)d_in[7];
    const float* chain = (const float*)d_in[8];
    float* out = (float*)d_out;

    char* ws = (char*)d_ws;
    size_t off = 0;
    auto alloc = [&](size_t bytes) { size_t o = off; off = (off + bytes + 255) & ~(size_t)255; return o; };

    unsigned short* xb  = (unsigned short*)(ws + alloc((size_t)2 * BN * DIM));
    unsigned short* WT0 = (unsigned short*)(ws + alloc((size_t)2 * 3 * DIM * DIM));
    unsigned short* WT1 = (unsigned short*)(ws + alloc((size_t)2 * 3 * DIM * DIM));
    unsigned short* WpT = (unsigned short*)(ws + alloc((size_t)2 * DIM * DIM));
    unsigned short* q0  = (unsigned short*)(ws + alloc((size_t)2 * BH * NN1 * DK));
    unsigned short* k0  = (unsigned short*)(ws + alloc((size_t)2 * BH * NN1 * DK));
    unsigned short* v0T = (unsigned short*)(ws + alloc((size_t)2 * BH * NN1 * DK));
    unsigned short* q1  = (unsigned short*)(ws + alloc((size_t)2 * BH * NN1 * DK));
    unsigned short* k1  = (unsigned short*)(ws + alloc((size_t)2 * BH * NN1 * DK));
    unsigned short* v1T = (unsigned short*)(ws + alloc((size_t)2 * BH * NN1 * DK));
    float* S0 = (float*)(ws + alloc(4 * (size_t)BH * NSQ));
    float* S1 = (float*)(ws + alloc(4 * (size_t)BH * NSQ));
    unsigned short* A0  = (unsigned short*)(ws + alloc(2 * (size_t)BH * NSQ));
    unsigned short* A0T = (unsigned short*)(ws + alloc(2 * (size_t)BH * NSQ));
    unsigned short* A1  = (unsigned short*)(ws + alloc(2 * (size_t)BH * NSQ));
    unsigned short* A1T = (unsigned short*)(ws + alloc(2 * (size_t)BH * NSQ));
    float* Cf = (float*)(ws + alloc(4 * (size_t)BH * NSQ));
    float* Cb = (float*)(ws + alloc(4 * (size_t)BH * NSQ));   // Smix overwrites in place
    unsigned short* Am = (unsigned short*)(ws + alloc(2 * (size_t)BH * NSQ));
    unsigned short* tT = (unsigned short*)(ws + alloc((size_t)2 * BH * NN1 * DK));
    unsigned short* yb = (unsigned short*)(ws + alloc((size_t)2 * BN * DIM));

    const int TB = 256;
    auto blocksForWaves = [](int waves) { return (waves + 7) / 8; };

    // 1) prep
    {
        int n = BN * DIM;
        k_cast_bf16<<<(n + TB - 1) / TB, TB, 0, stream>>>(x, xb, n);
        int nw = DIM * 3 * DIM;
        k_transpose_bf16<<<(nw + TB - 1) / TB, TB, 0, stream>>>(Wq0, WT0, DIM, 3 * DIM);
        k_transpose_bf16<<<(nw + TB - 1) / TB, TB, 0, stream>>>(Wq1, WT1, DIM, 3 * DIM);
        int np = DIM * DIM;
        k_transpose_bf16<<<(np + TB - 1) / TB, TB, 0, stream>>>(Wp, WpT, DIM, DIM);
    }
    // 2) QKV
    {
        int waves = 2 * (BN / 16) * ((3 * DIM) / 16);
        k_qkv<<<blocksForWaves(waves), TB, 0, stream>>>(xb, WT0, WT1, q0, k0, v0T, q1, k1, v1T);
    }
    // 3) scores S0, S1 (scale folded into q)
    {
        int waves = BH * (NN1 / 32) * (NN1 / 64);
        k_gemm_bf16_f32<<<blocksForWaves(waves), TB, 0, stream>>>(
            q0, k0, S0, NN1, NN1, DK, BH, (size_t)NN1 * DK, (size_t)NN1 * DK, NSQ);
        k_gemm_bf16_f32<<<blocksForWaves(waves), TB, 0, stream>>>(
            q1, k1, S1, NN1, NN1, DK, BH, (size_t)NN1 * DK, (size_t)NN1 * DK, NSQ);
    }
    // 4) softmaxes -> A0/A0T, A1/A1T
    k_softmax<<<BH * NN1, TB, 0, stream>>>(S0, A0, A0T);
    k_softmax<<<BH * NN1, TB, 0, stream>>>(S1, A1, A1T);
    // 5) C_fwd = A0@A1, C_bwd = A1@A0   (the N^3 GEMMs, 2x4 blocked WMMA)
    {
        int waves = BH * (NN1 / 32) * (NN1 / 64);
        k_gemm_bf16_f32<<<blocksForWaves(waves), TB, 0, stream>>>(
            A0, A1T, Cf, NN1, NN1, NN1, BH, NSQ, NSQ, NSQ);
        k_gemm_bf16_f32<<<blocksForWaves(waves), TB, 0, stream>>>(
            A1, A0T, Cb, NN1, NN1, NN1, BH, NSQ, NSQ, NSQ);
    }
    // 6) fused gates + Smix (Smix -> Cb in place)
    k_gates<<<BH * 32 * 32, TB, 0, stream>>>(S0, S1, Cf, Cb, c1w, c1b, c2w, c2b);
    // 7) softmax(Smix) -> Am
    k_softmax<<<BH * NN1, TB, 0, stream>>>(Cb, Am, nullptr);
    // 8) t = A1 @ v1 (transposed out)
    {
        int waves = BH * (NN1 / 16);
        k_tchain<<<blocksForWaves(waves), TB, 0, stream>>>(A1, v1T, tT);
    }
    // 9) y = Am@v0 + sig*A0@t
    {
        int waves = BH * (NN1 / 16);
        k_y<<<blocksForWaves(waves), TB, 0, stream>>>(Am, v0T, A0, tT, chain, yb);
    }
    // 10) out = y @ Wproj (f32 out)
    {
        int waves = (BN / 32) * (DIM / 64);
        k_gemm_bf16_f32<<<blocksForWaves(waves), TB, 0, stream>>>(
            yb, WpT, out, BN, DIM, DIM, 1, 0, 0, 0);
    }
}
